// TrajReconAffordanceNNDist_44830868635884
// MI455X (gfx1250) — compile-verified, matrix operand order checked by reading, not run on hardware
//
#include <hip/hip_runtime.h>
#include <cstdint>
#include <cstddef>

#define USE_ASYNC 1

// ---------------------------------------------------------------------------
// Types for CDNA5 WMMA (wave32)
// ---------------------------------------------------------------------------
typedef __attribute__((ext_vector_type(16))) __bf16 bf16x16;
typedef __attribute__((ext_vector_type(8)))  float  f32x8;

#define GEMM_BM 64
#define GEMM_BN 64
#define GEMM_BK 32

// ---------------------------------------------------------------------------
// Tiled GEMM: C[M,N] = act(A[M,Kp] @ Wt[N,Kp]^T + bias)
// A: bf16, row-major, K padded to Kp (multiple of 32, zero-filled pad).
// Wt: bf16, [Npad][Kp] (weight transposed + zero-padded rows/cols).
// C: f32. act: 0 = none, 1 = relu, 2 = leaky(0.01).
// 128 threads = 4 waves; wave w owns rows [16w,16w+16) of the 64x64 tile.
// Staging: each thread moves four 16-byte chunks per K-tile via
// global_load_async_to_lds_b128 (ASYNCcnt), fully branch-free (A rows clamped;
// their C rows are masked at the store).
// ---------------------------------------------------------------------------
__global__ __launch_bounds__(128) void gemm_bf16_wmma(
    const __bf16* __restrict__ A, const __bf16* __restrict__ Wt,
    const float* __restrict__ bias, float* __restrict__ C,
    int M, int Kp, int N, int act)
{
  __shared__ __bf16 sA[GEMM_BM][GEMM_BK];   // 4 KB
  __shared__ __bf16 sB[GEMM_BN][GEMM_BK];   // 4 KB

  const int tid  = threadIdx.x;
  const int wave = tid >> 5;
  const int lane = tid & 31;
  const int m0 = blockIdx.x * GEMM_BM;
  const int n0 = blockIdx.y * GEMM_BN;

  // Copy-chunk coords: chunk = (row cr, 8-element group cq); 2 chunks/matrix/thread.
  const int cr = tid >> 2;            // 0..31
  const int cq = (tid & 3) * 8;       // 0,8,16,24
  int gmA0 = m0 + cr;      if (gmA0 >= M) gmA0 = M - 1;   // clamp: C-store masks rows >= M
  int gmA1 = m0 + cr + 32; if (gmA1 >= M) gmA1 = M - 1;
  const int gnB0 = n0 + cr;           // Wt padded to Npad rows: always in range
  const int gnB1 = n0 + cr + 32;

#if USE_ASYNC
  const unsigned la0 = (unsigned)(unsigned long long)&sA[cr][cq];
  const unsigned la1 = (unsigned)(unsigned long long)&sA[cr + 32][cq];
  const unsigned lb0 = (unsigned)(unsigned long long)&sB[cr][cq];
  const unsigned lb1 = (unsigned)(unsigned long long)&sB[cr + 32][cq];
#endif

  f32x8 acc[4] = {};   // four 16x16 col-subtiles

  for (int k0 = 0; k0 < Kp; k0 += GEMM_BK) {
    const __bf16* ga0 = A  + (size_t)gmA0 * Kp + k0 + cq;   // 16-B aligned (Kp%32==0)
    const __bf16* ga1 = A  + (size_t)gmA1 * Kp + k0 + cq;
    const __bf16* gb0 = Wt + (size_t)gnB0 * Kp + k0 + cq;
    const __bf16* gb1 = Wt + (size_t)gnB1 * Kp + k0 + cq;
#if USE_ASYNC
    asm volatile(
        "global_load_async_to_lds_b128 %0, %4, off\n\t"
        "global_load_async_to_lds_b128 %1, %5, off\n\t"
        "global_load_async_to_lds_b128 %2, %6, off\n\t"
        "global_load_async_to_lds_b128 %3, %7, off\n\t"
        "s_wait_asynccnt 0x0"
        :
        : "v"(la0), "v"(la1), "v"(lb0), "v"(lb1),
          "v"(ga0), "v"(ga1), "v"(gb0), "v"(gb1)
        : "memory");
#else
    *(uint4*)&sA[cr][cq]      = *(const uint4*)ga0;
    *(uint4*)&sA[cr + 32][cq] = *(const uint4*)ga1;
    *(uint4*)&sB[cr][cq]      = *(const uint4*)gb0;
    *(uint4*)&sB[cr + 32][cq] = *(const uint4*)gb1;
#endif
    __syncthreads();

    // A fragment per ISA 16-bit A 16x32 layout:
    //   lanes 0-15 : M=lane,    elems 0..7 -> K=0..7,  elems 8..15 -> K=16..23
    //   lanes 16-31: M=lane-16, elems 0..7 -> K=8..15, elems 8..15 -> K=24..31
    const int mrow  = lane & 15;
    const int khalf = (lane >> 4) * 8;
    bf16x16 afrag;
#pragma unroll
    for (int e = 0; e < 8; ++e) {
      afrag[e]     = sA[16 * wave + mrow][khalf + e];
      afrag[8 + e] = sA[16 * wave + mrow][khalf + 16 + e];
    }
    bf16x16 bfrag0, bfrag1, bfrag2, bfrag3;   // symmetric layout, N in place of M
#pragma unroll
    for (int e = 0; e < 8; ++e) {
      bfrag0[e]     = sB[ 0 + mrow][khalf + e];
      bfrag0[8 + e] = sB[ 0 + mrow][khalf + 16 + e];
      bfrag1[e]     = sB[16 + mrow][khalf + e];
      bfrag1[8 + e] = sB[16 + mrow][khalf + 16 + e];
      bfrag2[e]     = sB[32 + mrow][khalf + e];
      bfrag2[8 + e] = sB[32 + mrow][khalf + 16 + e];
      bfrag3[e]     = sB[48 + mrow][khalf + e];
      bfrag3[8 + e] = sB[48 + mrow][khalf + 16 + e];
    }
    acc[0] = __builtin_amdgcn_wmma_f32_16x16x32_bf16(
        false, afrag, false, bfrag0, (short)0, acc[0], true,  false);
    acc[1] = __builtin_amdgcn_wmma_f32_16x16x32_bf16(
        false, afrag, false, bfrag1, (short)0, acc[1], true,  false);
    acc[2] = __builtin_amdgcn_wmma_f32_16x16x32_bf16(
        false, afrag, false, bfrag2, (short)0, acc[2], true,  false);
    acc[3] = __builtin_amdgcn_wmma_f32_16x16x32_bf16(
        false, afrag, false, bfrag3, (short)0, acc[3], false, false);
    __syncthreads();
  }

  // C/D layout: VGPR p: lanes 0-15 -> (M=p, N=lane), lanes 16-31 -> (M=p+8, N=lane-16)
  const int coln = lane & 15;
  const int rofs = (lane >> 4) * 8;
#pragma unroll
  for (int j = 0; j < 4; ++j) {
#pragma unroll
    for (int p = 0; p < 8; ++p) {
      int gm = m0 + 16 * wave + rofs + p;
      int gn = n0 + 16 * j + coln;
      if (gm < M && gn < N) {
        float v = acc[j][p];
        if (bias) v += bias[gn];
        if (act == 1)      v = fmaxf(v, 0.0f);
        else if (act == 2) v = (v > 0.0f) ? v : 0.01f * v;
        C[(size_t)gm * N + gn] = v;
      }
    }
  }
}

// Weight prep: W[K,N] f32 -> Wt[Npad][Kp] bf16 (transposed, zero-padded)
__global__ void wprep(const float* __restrict__ W, __bf16* __restrict__ Wt,
                      int K, int N, int Kp, int Np)
{
  int t = blockIdx.x * blockDim.x + threadIdx.x;
  if (t >= Np * Kp) return;
  int n = t / Kp, k = t % Kp;
  Wt[t] = (k < K && n < N) ? (__bf16)W[(size_t)k * N + n] : (__bf16)0.0f;
}

// f32 [M,C] -> bf16 [M,Kp] zero-padded
__global__ void cvt_bf16pad(const float* __restrict__ X, __bf16* __restrict__ Y,
                            long long M, int C, int Kp)
{
  long long tot = M * Kp;
  long long e = (long long)blockIdx.x * blockDim.x + threadIdx.x;
  if (e >= tot) return;
  int c = (int)(e % Kp);
  long long r = e / Kp;
  Y[e] = (c < C) ? (__bf16)X[r * C + c] : (__bf16)0.0f;
}

// ---------------------------------------------------------------------------
// Farthest point sampling: one block per batch. Points in LDS, distances in
// registers (8 per thread, 512 threads, n <= 4096), LDS tree argmax.
// ---------------------------------------------------------------------------
__global__ __launch_bounds__(512) void fps_kernel(
    const float* __restrict__ xyz, int n, int npoint, int* __restrict__ out)
{
  __shared__ float px[4096], py[4096], pz[4096];
  __shared__ float rv[512];
  __shared__ int   ri[512];

  const int b = blockIdx.x, tid = threadIdx.x;
  const float* base = xyz + (size_t)b * n * 3;

  float dist[8];
#pragma unroll
  for (int j = 0; j < 8; ++j) dist[j] = 1e10f;

  for (int i = tid; i < n; i += 512) {
    px[i] = base[3 * i]; py[i] = base[3 * i + 1]; pz[i] = base[3 * i + 2];
  }
  __syncthreads();

  int last = 0;
  for (int s = 0; s < npoint; ++s) {
    if (tid == 0) out[(size_t)b * npoint + s] = last;
    float lx = px[last], ly = py[last], lz = pz[last];
    float bv = -1.0f; int bi = 0x7fffffff;
#pragma unroll
    for (int j = 0; j < 8; ++j) {
      int i = tid + 512 * j;
      if (i < n) {
        float dx = px[i] - lx, dy = py[i] - ly, dz = pz[i] - lz;
        float d  = dx * dx + dy * dy + dz * dz;
        float dm = fminf(dist[j], d);
        dist[j] = dm;
        if (dm > bv || (dm == bv && i < bi)) { bv = dm; bi = i; }
      }
    }
    rv[tid] = bv; ri[tid] = bi;
    __syncthreads();
    for (int off = 256; off > 0; off >>= 1) {
      if (tid < off) {
        float v2 = rv[tid + off]; int i2 = ri[tid + off];
        if (v2 > rv[tid] || (v2 == rv[tid] && i2 < ri[tid])) { rv[tid] = v2; ri[tid] = i2; }
      }
      __syncthreads();
    }
    last = ri[0];
    __syncthreads();
  }
}

// ---------------------------------------------------------------------------
__global__ void gather_centers(const float* __restrict__ xyz, const int* __restrict__ ci,
                               float* __restrict__ out, int B_, int np, int n)
{
  int t = blockIdx.x * blockDim.x + threadIdx.x;
  if (t >= B_ * np * 3) return;
  int d = t % 3, rc = t / 3, b = rc / np, c = rc % np;
  int id = ci[b * np + c];
  out[t] = xyz[((size_t)b * n + id) * 3 + d];
}

// Ball query: first `ns` indices (ascending) with d2 < r2; pad with first hit,
// or (n-1) if no hit.
__global__ void ball_query(const float* __restrict__ centers, const float* __restrict__ xyz,
                           int B_, int np, int n, float r2, int ns, int* __restrict__ out)
{
  int t = blockIdx.x * blockDim.x + threadIdx.x;
  if (t >= B_ * np) return;
  int b = t / np;
  const float* ctr = centers + (size_t)t * 3;
  float cx = ctr[0], cy = ctr[1], cz = ctr[2];
  const float* base = xyz + (size_t)b * n * 3;
  int* o = out + (size_t)t * ns;
  int cnt = 0;
  for (int i = 0; i < n && cnt < ns; ++i) {
    float dx = base[3 * i] - cx, dy = base[3 * i + 1] - cy, dz = base[3 * i + 2] - cz;
    if (dx * dx + dy * dy + dz * dz < r2) o[cnt++] = i;
  }
  int pad = (cnt > 0) ? o[0] : (n - 1);
  for (int i = cnt; i < ns; ++i) o[i] = pad;
}

// Build grouped rows directly as bf16 GEMM input (K padded to Kp):
// G[(b,c,s), :] = [xyz[gi]-center (3) | feats[gi] (Cf) | 0-pad]
__global__ void group_concat_bf16(const float* __restrict__ xyz, const float* __restrict__ centers,
                                  const float* __restrict__ feats, const int* __restrict__ gi,
                                  int B_, int np, int n, int ns, int Cf, int Kp,
                                  __bf16* __restrict__ G)
{
  int Wd = 3 + Cf;
  long long total = (long long)B_ * np * ns * Kp;
  long long e = (long long)blockIdx.x * blockDim.x + threadIdx.x;
  if (e >= total) return;
  long long row = e / Kp; int ch = (int)(e % Kp);
  float v = 0.0f;
  if (ch < Wd) {
    int b = (int)(row / ((long long)np * ns));
    int rem = (int)(row % ((long long)np * ns));
    int c = rem / ns, s = rem % ns;
    int id = gi[((size_t)b * np + c) * ns + s];
    if (ch < 3) v = xyz[((size_t)b * n + id) * 3 + ch] - centers[((size_t)b * np + c) * 3 + ch];
    else        v = feats[((size_t)b * n + id) * Cf + (ch - 3)];
  }
  G[e] = (__bf16)v;
}

// ---------------------------------------------------------------------------
// BatchNorm (training mode): two-pass over all M rows.
// ---------------------------------------------------------------------------
__global__ __launch_bounds__(256) void bn_stats(const float* __restrict__ X, int M, int C,
                                                float* __restrict__ stats /* [sum|sumsq] */)
{
  int r0 = blockIdx.x * 256;
  int r1 = (r0 + 256 < M) ? (r0 + 256) : M;
  for (int c = threadIdx.x; c < C; c += 256) {
    float s = 0.0f, ss = 0.0f;
    for (int r = r0; r < r1; ++r) {
      float v = X[(size_t)r * C + c];
      s += v; ss += v * v;
    }
    atomicAdd(&stats[c], s);
    atomicAdd(&stats[C + c], ss);
  }
}

// BN+ReLU, f32 out (for maxpool / lfnew / stats-consumers)
__global__ void bn_apply_relu(const float* __restrict__ X, float* __restrict__ Y,
                              const float* __restrict__ stats, const float* __restrict__ gamma,
                              const float* __restrict__ beta, long long M, int C)
{
  long long tot = M * C;
  long long e = (long long)blockIdx.x * blockDim.x + threadIdx.x;
  if (e >= tot) return;
  int c = (int)(e % C);
  float inv  = 1.0f / (float)M;
  float mean = stats[c] * inv;
  float var  = stats[C + c] * inv - mean * mean;
  float v = gamma[c] * (X[e] - mean) * rsqrtf(var + 1e-5f) + beta[c];
  Y[e] = fmaxf(v, 0.0f);
}

// BN+ReLU producing the next GEMM's bf16 K-padded input
__global__ void bn_apply_relu_bf16(const float* __restrict__ X, __bf16* __restrict__ Y,
                                   const float* __restrict__ stats, const float* __restrict__ gamma,
                                   const float* __restrict__ beta, long long M, int C, int Kp)
{
  long long tot = M * Kp;
  long long e = (long long)blockIdx.x * blockDim.x + threadIdx.x;
  if (e >= tot) return;
  int c = (int)(e % Kp);
  long long r = e / Kp;
  float v = 0.0f;
  if (c < C) {
    float inv  = 1.0f / (float)M;
    float mean = stats[c] * inv;
    float var  = stats[C + c] * inv - mean * mean;
    v = gamma[c] * (X[r * C + c] - mean) * rsqrtf(var + 1e-5f) + beta[c];
    v = fmaxf(v, 0.0f);
  }
  Y[e] = (__bf16)v;
}

// BN+ReLU applied only to point-0 row of each batch (for f_s)
__global__ void bn_rows0(const float* __restrict__ Y, const float* __restrict__ stats,
                         const float* __restrict__ g, const float* __restrict__ bt,
                         int M, int C, int stride_rows, int B_, float* __restrict__ out)
{
  int t = blockIdx.x * blockDim.x + threadIdx.x;
  if (t >= B_ * C) return;
  int b = t / C, c = t % C;
  float inv  = 1.0f / (float)M;
  float mean = stats[c] * inv;
  float var  = stats[C + c] * inv - mean * mean;
  float v = Y[((size_t)b * stride_rows) * C + c];
  v = g[c] * (v - mean) * rsqrtf(var + 1e-5f) + bt[c];
  out[t] = fmaxf(v, 0.0f);
}

// Max over the nsample axis: X (R, ns, C) -> Y (R, C)
__global__ void maxpool_ns(const float* __restrict__ X, float* __restrict__ Y,
                           int R, int ns, int C)
{
  long long tot = (long long)R * C;
  long long e = (long long)blockIdx.x * blockDim.x + threadIdx.x;
  if (e >= tot) return;
  long long r = e / C; int c = (int)(e % C);
  float m = -3.4e38f;
  for (int s = 0; s < ns; ++s)
    m = fmaxf(m, X[((size_t)r * ns + s) * C + c]);
  Y[e] = m;
}

// ---------------------------------------------------------------------------
// FP: 3 nearest neighbors + inverse-distance weights
// ---------------------------------------------------------------------------
__global__ void knn3(const float* __restrict__ ux, const float* __restrict__ kx,
                     int B_, int nu, int nk, int* __restrict__ idx3, float* __restrict__ w3)
{
  int t = blockIdx.x * blockDim.x + threadIdx.x;
  if (t >= B_ * nu) return;
  int b = t / nu;
  float x = ux[(size_t)t * 3], y = ux[(size_t)t * 3 + 1], z = ux[(size_t)t * 3 + 2];
  const float* base = kx + (size_t)b * nk * 3;
  float d0 = 1e30f, d1 = 1e30f, d2 = 1e30f;
  int i0 = 0, i1 = 0, i2 = 0;
  for (int j = 0; j < nk; ++j) {
    float dx = base[3 * j] - x, dy = base[3 * j + 1] - y, dz = base[3 * j + 2] - z;
    float d = dx * dx + dy * dy + dz * dz;
    if (d < d0)      { d2 = d1; i2 = i1; d1 = d0; i1 = i0; d0 = d; i0 = j; }
    else if (d < d1) { d2 = d1; i2 = i1; d1 = d; i1 = j; }
    else if (d < d2) { d2 = d; i2 = j; }
  }
  float w0 = 1.0f / (d0 + 1e-8f), w1 = 1.0f / (d1 + 1e-8f), w2 = 1.0f / (d2 + 1e-8f);
  float s = 1.0f / (w0 + w1 + w2);
  idx3[(size_t)t * 3]     = i0; w3[(size_t)t * 3]     = w0 * s;
  idx3[(size_t)t * 3 + 1] = i1; w3[(size_t)t * 3 + 1] = w1 * s;
  idx3[(size_t)t * 3 + 2] = i2; w3[(size_t)t * 3 + 2] = w2 * s;
}

// X[(b,i), :] = [ sum_t w*kfeat[idx_t] (Ck) | skip (Cs) | 0-pad ], bf16 K-padded
__global__ void fp_interp_concat_bf16(const float* __restrict__ kfeat, const float* __restrict__ skip,
                                      const int* __restrict__ idx3, const float* __restrict__ w3,
                                      int B_, int nu, int nk, int Ck, int Cs, int Kp,
                                      __bf16* __restrict__ X)
{
  int Wd = Ck + Cs;
  long long tot = (long long)B_ * nu * Kp;
  long long e = (long long)blockIdx.x * blockDim.x + threadIdx.x;
  if (e >= tot) return;
  long long row = e / Kp; int ch = (int)(e % Kp);
  float v = 0.0f;
  if (ch < Ck) {
    int b = (int)(row / nu);
    for (int t = 0; t < 3; ++t) {
      int id = idx3[row * 3 + t];
      v += w3[row * 3 + t] * kfeat[((size_t)b * nk + id) * Ck + ch];
    }
  } else if (ch < Wd) {
    v = skip[row * Cs + (ch - Ck)];
  }
  X[e] = (__bf16)v;
}

// ---------------------------------------------------------------------------
// Head helpers
// ---------------------------------------------------------------------------
__global__ void concat3_bf16(const float* __restrict__ a, int ca, const float* __restrict__ b, int cb,
                             const float* __restrict__ c, int cc, int R, int Kp,
                             __bf16* __restrict__ o)
{
  int Wd = ca + cb + cc;
  int t = blockIdx.x * blockDim.x + threadIdx.x;
  if (t >= R * Kp) return;
  int r = t / Kp, ch = t % Kp;
  float v = 0.0f;
  if (ch < ca)           v = a[(size_t)r * ca + ch];
  else if (ch < ca + cb) v = b[(size_t)r * cb + (ch - ca)];
  else if (ch < Wd)      v = c[(size_t)r * cc + (ch - ca - cb)];
  o[t] = (__bf16)v;
}

__global__ void z_kernel(const float* __restrict__ mu, const float* __restrict__ lv,
                         const float* __restrict__ noise, float* __restrict__ z,
                         float* __restrict__ out_mu, float* __restrict__ out_lv, int n)
{
  int t = blockIdx.x * blockDim.x + threadIdx.x;
  if (t >= n) return;
  float m = mu[t], l = lv[t];
  out_mu[t] = m;
  out_lv[t] = l;
  z[t] = m + __expf(l * 0.5f) * noise[t];
}

// ---------------------------------------------------------------------------
// Host orchestration
// ---------------------------------------------------------------------------
extern "C" void kernel_launch(void* const* d_in, const int* in_sizes, int n_in,
                              void* d_out, int out_size, void* d_ws, size_t ws_size,
                              hipStream_t stream)
{
  (void)in_sizes; (void)n_in; (void)out_size; (void)ws_size;

  // ----- parameter unpacking (setup_inputs dict insertion order) -----
  int pi = 4;
  auto nxtp = [&]() { return (const float*)d_in[pi++]; };

  const float *saW[4][3], *saG[4][3], *saB[4][3];
  for (int l = 0; l < 4; ++l)
    for (int k = 0; k < 3; ++k) { saW[l][k] = nxtp(); saG[l][k] = nxtp(); saB[l][k] = nxtp(); }

  const int fpn[4] = {3, 2, 2, 2};
  const float *fpW[4][3], *fpG[4][3], *fpB[4][3];
  for (int l = 0; l < 4; ++l)
    for (int k = 0; k < fpn[l]; ++k) { fpW[l][k] = nxtp(); fpG[l][k] = nxtp(); fpB[l][k] = nxtp(); }

  const float *fcW = nxtp(), *fcG = nxtp(), *fcB = nxtp();
  const float *cpW = nxtp(), *cpB = nxtp();
  const float *tW[3], *tB[3];
  for (int k = 0; k < 3; ++k) { tW[k] = nxtp(); tB[k] = nxtp(); }
  const float *eW[4], *eB[4];                      // mlp1, mlp2, mu, logvar
  for (int k = 0; k < 4; ++k) { eW[k] = nxtp(); eB[k] = nxtp(); }
  const float *dW[3], *dB[3];
  for (int k = 0; k < 3; ++k) { dW[k] = nxtp(); dB[k] = nxtp(); }

  // ----- dims -----
  const int n_arr[5]  = {4096, 1024, 256, 64, 16};
  const int np_arr[4] = {1024, 256, 64, 16};
  const float r2_arr[4] = {0.01f, 0.04f, 0.16f, 0.64f};
  const int Cf[5]  = {3, 64, 128, 256, 512};
  const int sad[4][4] = {{6,32,32,64},{67,64,64,128},{131,128,128,256},{259,256,256,512}};
  const int fpd[4][4] = {{131,128,128,128},{320,256,128,0},{384,256,256,0},{768,256,256,0}};

  auto kpad = [](int K) { return (K + 31) & ~31; };

  // ----- workspace bump allocator -----
  char* wsb = (char*)d_ws;
  size_t off = 0;
  auto alloc = [&](size_t bytes) -> void* {
    void* p = wsb + off;
    off = (off + bytes + 255) & ~(size_t)255;
    return p;
  };

  // ----- weight prep (f32 [K,N] -> bf16 [Npad][Kp], once per launch) -----
  auto wprep_run = [&](const float* W, int K, int N) -> const __bf16* {
    int Kp = kpad(K), Np = (N + 63) & ~63;
    __bf16* Wt = (__bf16*)alloc((size_t)Np * Kp * 2);
    int tot = Np * Kp;
    wprep<<<(tot + 255) / 256, 256, 0, stream>>>(W, Wt, K, N, Kp, Np);
    return Wt;
  };
  const __bf16 *saWt[4][3], *fpWt[4][3], *fcWt, *cpWt, *tWt[3], *eWt[4], *dWt[3];
  for (int l = 0; l < 4; ++l)
    for (int k = 0; k < 3; ++k) saWt[l][k] = wprep_run(saW[l][k], sad[l][k], sad[l][k + 1]);
  for (int l = 0; l < 4; ++l)
    for (int k = 0; k < fpn[l]; ++k) fpWt[l][k] = wprep_run(fpW[l][k], fpd[l][k], fpd[l][k + 1]);
  fcWt = wprep_run(fcW, 128, 256);
  cpWt = wprep_run(cpW, 3, 32);
  tWt[0] = wprep_run(tW[0], 180, 256);
  tWt[1] = wprep_run(tW[1], 256, 256);
  tWt[2] = wprep_run(tW[2], 256, 128);
  eWt[0] = wprep_run(eW[0], 416, 128);
  eWt[1] = wprep_run(eW[1], 128, 64);
  eWt[2] = wprep_run(eW[2], 64, 64);
  eWt[3] = wprep_run(eW[3], 64, 64);
  dWt[0] = wprep_run(dW[0], 352, 128);
  dWt[1] = wprep_run(dW[1], 128, 128);
  dWt[2] = wprep_run(dW[2], 128, 180);

  // ----- big buffers -----
  const float* xyzL[5];
  xyzL[0] = (const float*)d_in[0];
  float* xyzLw[5] = {nullptr, nullptr, nullptr, nullptr, nullptr};
  for (int i = 1; i <= 4; ++i) {
    xyzLw[i] = (float*)alloc((size_t)16 * np_arr[i - 1] * 3 * 4);
    xyzL[i]  = xyzLw[i];
  }
  const float* featL[5];
  featL[0] = (const float*)d_in[0];
  float* featLw[5] = {nullptr, nullptr, nullptr, nullptr, nullptr};
  for (int i = 1; i <= 4; ++i) {
    featLw[i] = (float*)alloc((size_t)16 * np_arr[i - 1] * Cf[i] * 4);
    featL[i]  = featLw[i];
  }
  float* lfnew[4];
  const int lfnewC[4] = {128, 128, 256, 256};
  for (int i = 0; i < 4; ++i)
    lfnew[i] = (float*)alloc((size_t)16 * n_arr[i] * lfnewC[i] * 4);

  int*   ci   = (int*)alloc((size_t)16 * 1024 * 4);
  int*   gi   = (int*)alloc((size_t)16 * 1024 * 32 * 4);
  int*   idx3 = (int*)alloc((size_t)16 * 4096 * 3 * 4);
  float* w3   = (float*)alloc((size_t)16 * 4096 * 3 * 4);
  __bf16* bfP = (__bf16*)alloc((size_t)524288 * 32 * 2);  // 33.5 MB bf16 ping
  __bf16* bfQ = (__bf16*)alloc((size_t)524288 * 32 * 2);  // 33.5 MB bf16 pong
  float* bufY = (float*)alloc((size_t)524288 * 64 * 4);   // 134 MB f32 GEMM out
  float* stats = (float*)alloc(2 * 512 * 4);

  float* f_s    = (float*)alloc(16 * 256 * 4);
  float* f_cp   = (float*)alloc(16 * 32 * 4);
  float* tb0    = (float*)alloc(16 * 256 * 4);
  float* tb1    = (float*)alloc(16 * 256 * 4);
  float* tb2    = (float*)alloc(16 * 128 * 4);
  float* net1   = (float*)alloc(16 * 128 * 4);
  float* net2   = (float*)alloc(16 * 64 * 4);
  float* mu_b   = (float*)alloc(16 * 64 * 4);
  float* lv_b   = (float*)alloc(16 * 64 * 4);
  float* zbuf   = (float*)alloc(16 * 64 * 4);
  float* h1     = (float*)alloc(16 * 128 * 4);
  float* h2     = (float*)alloc(16 * 128 * 4);

  __bf16* bf_cp   = (__bf16*)alloc(16 * 32 * 2);
  __bf16* bf_traj = (__bf16*)alloc(16 * 192 * 2);
  __bf16* bf_t0   = (__bf16*)alloc(16 * 256 * 2);
  __bf16* bf_t1   = (__bf16*)alloc(16 * 256 * 2);
  __bf16* bf_enc  = (__bf16*)alloc(16 * 416 * 2);
  __bf16* bf_n1   = (__bf16*)alloc(16 * 128 * 2);
  __bf16* bf_n2   = (__bf16*)alloc(16 * 64 * 2);
  __bf16* bf_dec  = (__bf16*)alloc(16 * 352 * 2);
  __bf16* bf_h1   = (__bf16*)alloc(16 * 128 * 2);
  __bf16* bf_h2   = (__bf16*)alloc(16 * 128 * 2);

  auto run_gemm = [&](const __bf16* A, const __bf16* Wt, const float* bias, float* C,
                      long long M, int K, int N, int act) {
    int Kp = kpad(K);
    dim3 grid((unsigned)((M + GEMM_BM - 1) / GEMM_BM), (N + GEMM_BN - 1) / GEMM_BN, 1);
    gemm_bf16_wmma<<<grid, dim3(128), 0, stream>>>(A, Wt, bias, C, (int)M, Kp, N, act);
  };
  auto run_stats = [&](const float* X, long long M, int C) {
    (void)hipMemsetAsync(stats, 0, (size_t)2 * C * sizeof(float), stream);
    bn_stats<<<(int)((M + 255) / 256), 256, 0, stream>>>(X, (int)M, C, stats);
  };
  auto run_cvt = [&](const float* X, __bf16* Y, long long M, int C, int Kp) {
    long long tot = M * Kp;
    cvt_bf16pad<<<(int)((tot + 255) / 256), 256, 0, stream>>>(X, Y, M, C, Kp);
  };

  // ========================== SA levels ==========================
  for (int lev = 0; lev < 4; ++lev) {
    const int n = n_arr[lev], np = np_arr[lev];
    fps_kernel<<<16, 512, 0, stream>>>(xyzL[lev], n, np, ci);

    int t1 = 16 * np * 3;
    gather_centers<<<(t1 + 255) / 256, 256, 0, stream>>>(xyzL[lev], ci, xyzLw[lev + 1], 16, np, n);

    int t2 = 16 * np;
    ball_query<<<(t2 + 255) / 256, 256, 0, stream>>>(xyzL[lev + 1], xyzL[lev], 16, np, n,
                                                     r2_arr[lev], 32, gi);

    long long M = (long long)16 * np * 32;
    int Kp0 = kpad(3 + Cf[lev]);
    long long tot = M * Kp0;
    group_concat_bf16<<<(int)((tot + 255) / 256), 256, 0, stream>>>(
        xyzL[lev], xyzL[lev + 1], featL[lev], gi, 16, np, n, 32, Cf[lev], Kp0, bfP);

    const __bf16* curbf = bfP;
    __bf16* otherbf = bfQ;
    for (int k = 0; k < 3; ++k) {
      int K = sad[lev][k], N = sad[lev][k + 1];
      run_gemm(curbf, saWt[lev][k], nullptr, bufY, M, K, N, 0);
      run_stats(bufY, M, N);
      if (k < 2) {
        int Kpn = kpad(N);
        long long t3 = M * Kpn;
        bn_apply_relu_bf16<<<(int)((t3 + 255) / 256), 256, 0, stream>>>(
            bufY, otherbf, stats, saG[lev][k], saB[lev][k], M, N, Kpn);
        const __bf16* tmp = curbf; curbf = otherbf; otherbf = (__bf16*)tmp;
      } else {
        long long t3 = M * N;
        bn_apply_relu<<<(int)((t3 + 255) / 256), 256, 0, stream>>>(
            bufY, bufY, stats, saG[lev][k], saB[lev][k], M, N);
        long long R = (long long)16 * np;
        long long t4 = R * N;
        maxpool_ns<<<(int)((t4 + 255) / 256), 256, 0, stream>>>(bufY, featLw[lev + 1], (int)R, 32, N);
      }
    }
  }

  // ========================== FP levels ==========================
  for (int i = 3; i >= 0; --i) {
    const int nu = n_arr[i], nk = n_arr[i + 1];
    const float* kfeat = (i == 3) ? featL[4] : lfnew[i + 1];
    int Ck = (i == 3) ? 512 : fpd[i + 1][fpn[i + 1]];
    const float* skip = (i == 0) ? (const float*)d_in[0] : featL[i];
    int Cs = (i == 0) ? 3 : sad[i - 1][3];

    int tk = 16 * nu;
    knn3<<<(tk + 255) / 256, 256, 0, stream>>>(xyzL[i], xyzL[i + 1], 16, nu, nk, idx3, w3);

    long long R = (long long)16 * nu;
    int Kp0 = kpad(Ck + Cs);
    long long tot = R * Kp0;
    fp_interp_concat_bf16<<<(int)((tot + 255) / 256), 256, 0, stream>>>(
        kfeat, skip, idx3, w3, 16, nu, nk, Ck, Cs, Kp0, bfP);

    const __bf16* curbf = bfP;
    __bf16* otherbf = bfQ;
    for (int k = 0; k < fpn[i]; ++k) {
      int K = fpd[i][k], N = fpd[i][k + 1];
      run_gemm(curbf, fpWt[i][k], nullptr, bufY, R, K, N, 0);
      run_stats(bufY, R, N);
      if (k < fpn[i] - 1) {
        int Kpn = kpad(N);
        long long t3 = R * Kpn;
        bn_apply_relu_bf16<<<(int)((t3 + 255) / 256), 256, 0, stream>>>(
            bufY, otherbf, stats, fpG[i][k], fpB[i][k], R, N, Kpn);
        const __bf16* tmp = curbf; curbf = otherbf; otherbf = (__bf16*)tmp;
      } else {
        long long t3 = R * N;
        bn_apply_relu<<<(int)((t3 + 255) / 256), 256, 0, stream>>>(
            bufY, lfnew[i], stats, fpG[i][k], fpB[i][k], R, N);
      }
    }
  }

  // ========================== FC / f_s ==========================
  run_cvt(lfnew[0], bfP, 65536, 128, 128);
  run_gemm(bfP, fcWt, nullptr, bufY, 65536, 128, 256, 0);
  run_stats(bufY, 65536, 256);
  bn_rows0<<<(16 * 256 + 255) / 256, 256, 0, stream>>>(bufY, stats, fcG, fcB,
                                                       65536, 256, 4096, 16, f_s);

  // ========================== Head ==========================
  float* dout = (float*)d_out;
  run_cvt((const float*)d_in[2], bf_cp, 16, 3, 32);
  run_gemm(bf_cp, cpWt, cpB, f_cp, 16, 3, 32, 0);

  run_cvt((const float*)d_in[1], bf_traj, 16, 180, 192);
  run_gemm(bf_traj, tWt[0], tB[0], tb0, 16, 180, 256, 2);
  run_cvt(tb0, bf_t0, 16, 256, 256);
  run_gemm(bf_t0, tWt[1], tB[1], tb1, 16, 256, 256, 2);
  run_cvt(tb1, bf_t1, 16, 256, 256);
  run_gemm(bf_t1, tWt[2], tB[2], tb2, 16, 256, 128, 0);

  concat3_bf16<<<(16 * 416 + 255) / 256, 256, 0, stream>>>(f_s, 256, tb2, 128, f_cp, 32,
                                                           16, 416, bf_enc);
  run_gemm(bf_enc, eWt[0], eB[0], net1, 16, 416, 128, 2);
  run_cvt(net1, bf_n1, 16, 128, 128);
  run_gemm(bf_n1, eWt[1], eB[1], net2, 16, 128, 64, 0);
  run_cvt(net2, bf_n2, 16, 64, 64);
  run_gemm(bf_n2, eWt[2], eB[2], mu_b, 16, 64, 64, 0);
  run_gemm(bf_n2, eWt[3], eB[3], lv_b, 16, 64, 64, 0);

  z_kernel<<<(1024 + 255) / 256, 256, 0, stream>>>(mu_b, lv_b, (const float*)d_in[3], zbuf,
                                                   dout + 2880, dout + 2880 + 1024, 1024);

  concat3_bf16<<<(16 * 352 + 255) / 256, 256, 0, stream>>>(f_s, 256, f_cp, 32, zbuf, 64,
                                                           16, 352, bf_dec);
  run_gemm(bf_dec, dWt[0], dB[0], h1, 16, 352, 128, 2);
  run_cvt(h1, bf_h1, 16, 128, 128);
  run_gemm(bf_h1, dWt[1], dB[1], h2, 16, 128, 128, 2);
  run_cvt(h2, bf_h2, 16, 128, 128);
  run_gemm(bf_h2, dWt[2], dB[2], dout, 16, 128, 180, 0);   // recon -> d_out[0:2880]
}
